// SwinTransformerBlock_57870389346656
// MI455X (gfx1250) — compile-verified, matrix operand order checked by reading, not run on hardware
//
#include <hip/hip_runtime.h>

// ---------------- CDNA5 WMMA helpers (gfx1250, wave32) ----------------
typedef __attribute__((ext_vector_type(16))) _Float16 v16h;
typedef __attribute__((ext_vector_type(8)))  float    v8f;

constexpr int kDim   = 192;   // C
constexpr int kHd    = 32;    // head dim
constexpr int kWS    = 7;     // window size
constexpr int kNTok  = 49;    // tokens per window
constexpr int kShift = 3;
constexpr int kImg   = 56;
constexpr int kHid   = 768;   // MLP hidden
constexpr int kNWin  = 4096;  // 64 * 8 * 8
constexpr int kRows  = 200704;// 64*56*56 tokens

// k index inside a 16x32 f16 fragment for element e, lane half hf (ISA 7.12.2).
__device__ __forceinline__ int frag_k(int e, int hf) {
  int j = e >> 1, lo = e & 1;
  int base = (j < 4) ? (2 * j) : (16 + 2 * (j - 4));
  return base + 8 * hf + lo;
}

// Inverse: packed-tile offset contribution of k: hf*256 + e  (idx = tile*512 + rc*16 + pko(k))
__device__ __forceinline__ int pko(int k) {
  int hf = (k >> 3) & 1;
  int e  = 2 * (((k >> 4) ? 4 : 0) + ((k & 7) >> 1)) + (k & 1);
  return (hf << 8) + e;
}

__device__ __forceinline__ v16h ldfrag(const _Float16* p) {
  return *(const v16h*)p;   // 32B aligned -> 2x b128 loads
}

__device__ __forceinline__ v8f wmma16(v16h a, v16h b, v8f c) {
  return __builtin_amdgcn_wmma_f32_16x16x32_f16(false, a, false, b, (short)0, c, false, false);
}
__device__ __forceinline__ v8f vzero8() { v8f z = {}; return z; }

// ---------------- weight pack: fp32 (KxN row-major) -> f16 fragment tiles ----------------
// dst[tile*512 + lane*16 + e], tile = kt*(N/16)+nt ; B-fragment order (lane&15 = column).
__global__ void pack_w(const float* __restrict__ src, _Float16* __restrict__ dst, int K, int N) {
  int i = blockIdx.x * blockDim.x + threadIdx.x;
  if (i >= K * N) return;
  int tile = i >> 9, idx = i & 511;
  int lane = idx >> 4, e = idx & 15;
  int NT = N >> 4;
  int kt = tile / NT, nt = tile - kt * NT;
  int hf = lane >> 4, rc = lane & 15;
  int k = kt * 32 + frag_k(e, hf);
  int n = nt * 16 + rc;
  dst[i] = (_Float16)src[(size_t)k * N + n];
}

// ---------------- fused LN1 + shift + window attention ----------------
// One block per window: 96 threads = 3 waves; each wave handles 2 heads.
// Output written directly into globally fragment-packed A-tile layout:
//   attnP[((rg>>4)*6 + (c>>5))*512 + (rg&15)*16 + pko(c&31)]  (rg = flat token row, c = channel)
__global__ __launch_bounds__(96) void attn_kernel(
    const float* __restrict__ x, const float* __restrict__ g1, const float* __restrict__ b1,
    const _Float16* __restrict__ wqkv, const float* __restrict__ bqkv,
    _Float16* __restrict__ attnP) {
  __shared__ _Float16 AhP[24 * 512];         // packed A tiles [mi(4)][ks(6)], 24 KB
  __shared__ _Float16 wavebuf[3][3 * 2048];  // per wave: qP[2048] kP[2048] vP[2048]; pP overlays q+k
  __shared__ float s_mean[kNTok], s_rstd[kNTok];

  const int tid  = threadIdx.x;
  const int lane = tid & 31;
  const int wv   = tid >> 5;
  const int lofs = lane * 16;
  const int win  = blockIdx.x;
  const int b    = win >> 6;
  const int wh   = (win >> 3) & 7;
  const int ww   = win & 7;

  // ---- LN1 statistics (one thread per token) ----
  if (tid < kNTok) {
    int th = tid / kWS, tw = tid % kWS;
    int h0 = wh * kWS + th + kShift; if (h0 >= kImg) h0 -= kImg;
    int w0 = ww * kWS + tw + kShift; if (w0 >= kImg) w0 -= kImg;
    const float* row = x + ((size_t)(b * kImg + h0) * kImg + w0) * kDim;
    float s = 0.f, s2 = 0.f;
    for (int c = 0; c < kDim; ++c) { float v = row[c]; s += v; s2 += v * v; }
    float m = s / kDim;
    s_mean[tid] = m;
    s_rstd[tid] = rsqrtf(s2 / kDim - m * m + 1e-5f);
  }
  __syncthreads();

  // ---- normalize + gather straight into packed A-fragment layout, zero-pad rows 49..63 ----
  for (int i = tid; i < 64 * kDim; i += 96) {
    int t = i / kDim, c = i - t * kDim;
    _Float16 hv = (_Float16)0.f;
    if (t < kNTok) {
      int th = t / kWS, tw = t % kWS;
      int h0 = wh * kWS + th + kShift; if (h0 >= kImg) h0 -= kImg;
      int w0 = ww * kWS + tw + kShift; if (w0 >= kImg) w0 -= kImg;
      float v = x[((size_t)(b * kImg + h0) * kImg + w0) * kDim + c];
      hv = (_Float16)((v - s_mean[t]) * s_rstd[t] * g1[c] + b1[c]);
    }
    AhP[(((t >> 4) * 6 + (c >> 5)) << 9) + (t & 15) * 16 + pko(c & 31)] = hv;
  }
  __syncthreads();

  _Float16* qP = &wavebuf[wv][0];     // packed A tiles [mi(4)], k = head dim
  _Float16* kP = &wavebuf[wv][2048];  // packed B tiles [nt(4)] of K^T (k = head dim, n = token)
  _Float16* vP = &wavebuf[wv][4096];  // packed B tiles [kt(2)][nt(2)] (k = token, n = head dim)
  _Float16* pP = &wavebuf[wv][0];     // packed A tiles [mi(4)][kt(2)], overlays q+k

  const int hfD = lane >> 4;
  const int nlo = lane & 15;
  const float qscale = 0.1767766952966369f; // 32^-0.5 folded into q

  for (int hi = 0; hi < 2; ++hi) {
    const int head = wv * 2 + hi;

    // ---- QKV: (64x192) @ (192x32) per part ----
    for (int part = 0; part < 3; ++part) {
      const int colbase = part * kDim + head * kHd;
      const int cb16 = part * 12 + head * 2;  // colbase / 16
      v8f acc[4][2];
#pragma unroll
      for (int mi = 0; mi < 4; ++mi)
#pragma unroll
        for (int ni = 0; ni < 2; ++ni) acc[mi][ni] = vzero8();
      for (int ks = 0; ks < 6; ++ks) {
        v16h bfr[2];
#pragma unroll
        for (int ni = 0; ni < 2; ++ni)
          bfr[ni] = ldfrag(wqkv + (((size_t)ks * 36 + cb16 + ni) << 9) + lofs);
#pragma unroll
        for (int mi = 0; mi < 4; ++mi) {
          v16h af = ldfrag(AhP + ((mi * 6 + ks) << 9) + lofs);
#pragma unroll
          for (int ni = 0; ni < 2; ++ni) acc[mi][ni] = wmma16(af, bfr[ni], acc[mi][ni]);
        }
      }
      // epilogue: bias (+scale for q), scatter into packed per-wave tiles
      if (part < 2) {
        _Float16* dst = (part == 0) ? qP : kP;  // identical index map (tile=m>>4, rc=m&15, k=n)
        const float sc = (part == 0) ? qscale : 1.f;
#pragma unroll
        for (int ni = 0; ni < 2; ++ni) {
          int n = ni * 16 + nlo;
          float bias = bqkv[colbase + n];
          int c0 = pko(n);
#pragma unroll
          for (int mi = 0; mi < 4; ++mi)
#pragma unroll
            for (int r = 0; r < 8; ++r) {
              int m = mi * 16 + r + 8 * hfD;
              dst[((m >> 4) << 9) + (m & 15) * 16 + c0] = (_Float16)((acc[mi][ni][r] + bias) * sc);
            }
        }
      } else {
#pragma unroll
        for (int ni = 0; ni < 2; ++ni) {
          int n = ni * 16 + nlo;
          float bias = bqkv[colbase + n];
#pragma unroll
          for (int mi = 0; mi < 4; ++mi)
#pragma unroll
            for (int r = 0; r < 8; ++r) {
              int m = mi * 16 + r + 8 * hfD;   // token index = k of the PV GEMM
              vP[((((m >> 5) << 1) + ni) << 9) + nlo * 16 + pko(m & 31)] =
                  (_Float16)(acc[mi][ni][r] + bias);
            }
        }
      }
    }

    // ---- S = q @ k^T : load all fragments first (q/k LDS dead after; P overlays) ----
    v16h afr[4], bfr[4];
#pragma unroll
    for (int mi = 0; mi < 4; ++mi) afr[mi] = ldfrag(qP + (mi << 9) + lofs);
#pragma unroll
    for (int ni = 0; ni < 4; ++ni) bfr[ni] = ldfrag(kP + (ni << 9) + lofs);

    int c0p[4];
#pragma unroll
    for (int ni = 0; ni < 4; ++ni) c0p[ni] = pko(((ni & 1) << 4) + nlo);

    // ---- per 16-row strip: 4 WMMA + masked softmax + packed P scatter ----
#pragma unroll
    for (int mi = 0; mi < 4; ++mi) {
      v8f s[4];
#pragma unroll
      for (int ni = 0; ni < 4; ++ni) s[ni] = wmma16(afr[mi], bfr[ni], vzero8());
#pragma unroll
      for (int r = 0; r < 8; ++r) {
        float v0[4];
#pragma unroll
        for (int ni = 0; ni < 4; ++ni) {
          float v = s[ni][r];
          if (ni * 16 + nlo >= kNTok) v = -1e30f;
          v0[ni] = v;
        }
        float mx = fmaxf(fmaxf(v0[0], v0[1]), fmaxf(v0[2], v0[3]));
#pragma unroll
        for (int off = 1; off < 16; off <<= 1) mx = fmaxf(mx, __shfl_xor(mx, off, 32));
        float pn[4], sum = 0.f;
#pragma unroll
        for (int ni = 0; ni < 4; ++ni) { pn[ni] = expf(v0[ni] - mx); sum += pn[ni]; }
#pragma unroll
        for (int off = 1; off < 16; off <<= 1) sum += __shfl_xor(sum, off, 32);
        float inv = 1.f / sum;
        int m = mi * 16 + r + 8 * hfD;
        int base = ((m >> 4) << 10) + (m & 15) * 16;  // (m>>4)*2 tiles * 512
#pragma unroll
        for (int ni = 0; ni < 4; ++ni)
          pP[base + ((ni >> 1) << 9) + c0p[ni]] = (_Float16)(pn[ni] * inv);
      }
    }

    // ---- O = P @ V  (64x32, K=64 -> two WMMA steps) ----
    v8f ofr[4][2];
#pragma unroll
    for (int mi = 0; mi < 4; ++mi)
#pragma unroll
      for (int ni = 0; ni < 2; ++ni) ofr[mi][ni] = vzero8();
#pragma unroll
    for (int ks = 0; ks < 2; ++ks) {
      v16h bv[2];
#pragma unroll
      for (int ni = 0; ni < 2; ++ni) bv[ni] = ldfrag(vP + ((ks * 2 + ni) << 9) + lofs);
#pragma unroll
      for (int mi = 0; mi < 4; ++mi) {
        v16h ap = ldfrag(pP + ((mi * 2 + ks) << 9) + lofs);
#pragma unroll
        for (int ni = 0; ni < 2; ++ni) ofr[mi][ni] = wmma16(ap, bv[ni], ofr[mi][ni]);
      }
    }

    // ---- store valid rows straight into globally packed A-tile layout ----
    const int rgbase = win * kNTok;
#pragma unroll
    for (int ni = 0; ni < 2; ++ni) {
      int n = ni * 16 + nlo;           // channel within head -> c = head*32 + n, c>>5 == head
      int c0 = pko(n);
#pragma unroll
      for (int mi = 0; mi < 4; ++mi)
#pragma unroll
        for (int r = 0; r < 8; ++r) {
          int m = mi * 16 + r + 8 * hfD;
          if (m < kNTok) {
            int rg = rgbase + m;
            attnP[((size_t)((rg >> 4) * 6 + head) << 9) + (rg & 15) * 16 + c0] =
                (_Float16)ofr[mi][ni][r];
          }
        }
    }
  }
}

// ---------------- proj GEMM + window-reverse + un-shift + residual ----------------
// A block is a contiguous 6 KB packed tile -> async DMA to LDS (ASYNCcnt path).
__global__ __launch_bounds__(128) void proj_kernel(
    const _Float16* __restrict__ attnP, const _Float16* __restrict__ wproj,
    const float* __restrict__ bproj, const float* __restrict__ x, float* __restrict__ out) {
  __shared__ _Float16 AtP[6 * 512]; // packed A tiles [ks(6)], 6 KB
  const int tid = threadIdx.x, lane = tid & 31, wv = tid >> 5, lofs = lane * 16;
  const int blk = blockIdx.x;

  // gfx1250 async load to LDS: 128 lanes x 3 x 16B = 6144 B
  {
    const char* src = (const char*)(attnP + (size_t)blk * 3072) + tid * 16;
    unsigned int dst = (unsigned int)(size_t)(&AtP[0]) + (unsigned int)(tid * 16);
    asm volatile("global_load_async_to_lds_b128 %0, %1, off" :: "v"(dst), "v"(src) : "memory");
    asm volatile("global_load_async_to_lds_b128 %0, %1, off"
                 :: "v"(dst + 2048u), "v"(src + 2048) : "memory");
    asm volatile("global_load_async_to_lds_b128 %0, %1, off"
                 :: "v"(dst + 4096u), "v"(src + 4096) : "memory");
    asm volatile("s_wait_asynccnt 0x0" ::: "memory");
  }
  __syncthreads();

  v8f acc[3];
#pragma unroll
  for (int ni = 0; ni < 3; ++ni) acc[ni] = vzero8();
  for (int ks = 0; ks < 6; ++ks) {
    v16h af = ldfrag(AtP + (ks << 9) + lofs);
#pragma unroll
    for (int ni = 0; ni < 3; ++ni) {
      v16h bf = ldfrag(wproj + ((size_t)(ks * 12 + wv * 3 + ni) << 9) + lofs);
      acc[ni] = wmma16(af, bf, acc[ni]);
    }
  }

  const int hf = lane >> 4, nlo = lane & 15;
#pragma unroll
  for (int ni = 0; ni < 3; ++ni) {
    int col = wv * 48 + ni * 16 + nlo;
    float bias = bproj[col];
#pragma unroll
    for (int r = 0; r < 8; ++r) {
      int rg = blk * 16 + r + 8 * hf;
      int win = rg / kNTok, t = rg - win * kNTok;
      int b = win >> 6, wh = (win >> 3) & 7, ww = win & 7;
      int th = t / kWS, tw = t % kWS;
      int h0 = wh * kWS + th + kShift; if (h0 >= kImg) h0 -= kImg;
      int w0 = ww * kWS + tw + kShift; if (w0 >= kImg) w0 -= kImg;
      size_t xi = ((size_t)(b * kImg + h0) * kImg + w0) * kDim + col;
      out[xi] = acc[ni][r] + bias + x[xi]; // x_new = shortcut + proj(attn)
    }
  }
}

// ---------------- fused LN2 + fc1 + GELU + fc2 + residual ----------------
__global__ __launch_bounds__(128) void mlp_kernel(
    const float* __restrict__ g2, const float* __restrict__ b2,
    const _Float16* __restrict__ wfc1, const float* __restrict__ bfc1,
    const _Float16* __restrict__ wfc2, const float* __restrict__ bfc2,
    float* __restrict__ xio) {
  __shared__ _Float16 h2P[6 * 512];   // packed A tiles [ks(6)],  6 KB
  __shared__ _Float16 h1P[24 * 512];  // packed A tiles [ks(24)], 24 KB
  const int tid = threadIdx.x, lane = tid & 31, wv = tid >> 5, lofs = lane * 16;
  const size_t row0 = (size_t)blockIdx.x * 16;
  const int hf = lane >> 4, nlo = lane & 15;

  // ---- LN2: 8 consecutive lanes per row -> packed A layout ----
  {
    int r = tid >> 3, sub = tid & 7;
    const float* rp = xio + (row0 + r) * kDim;
    float s = 0.f, s2 = 0.f;
    for (int c = sub * 24; c < sub * 24 + 24; ++c) { float v = rp[c]; s += v; s2 += v * v; }
#pragma unroll
    for (int off = 1; off < 8; off <<= 1) { s += __shfl_xor(s, off, 32); s2 += __shfl_xor(s2, off, 32); }
    float m = s / kDim, rs = rsqrtf(s2 / kDim - m * m + 1e-5f);
    for (int c = sub * 24; c < sub * 24 + 24; ++c)
      h2P[((c >> 5) << 9) + r * 16 + pko(c & 31)] =
          (_Float16)((rp[c] - m) * rs * g2[c] + b2[c]);
  }
  __syncthreads();

  // ---- fc1 + exact GELU -> h1 (packed) ----
  {
    v8f acc[12];
#pragma unroll
    for (int ni = 0; ni < 12; ++ni) acc[ni] = vzero8();
    for (int ks = 0; ks < 6; ++ks) {
      v16h af = ldfrag(h2P + (ks << 9) + lofs);
#pragma unroll
      for (int ni = 0; ni < 12; ++ni) {
        v16h bf = ldfrag(wfc1 + ((size_t)(ks * 48 + wv * 12 + ni) << 9) + lofs);
        acc[ni] = wmma16(af, bf, acc[ni]);
      }
    }
#pragma unroll
    for (int ni = 0; ni < 12; ++ni) {
      int col = wv * kDim + ni * 16 + nlo;
      float bias = bfc1[col];
      int kt = wv * 6 + (ni >> 1);
      int c0 = pko(((ni & 1) << 4) + nlo);
#pragma unroll
      for (int r = 0; r < 8; ++r) {
        float v = acc[ni][r] + bias;
        v = 0.5f * v * (1.f + erff(v * 0.70710678118654752f)); // approximate=False
        h1P[(kt << 9) + (r + 8 * hf) * 16 + c0] = (_Float16)v;
      }
    }
  }
  __syncthreads();

  // ---- fc2 + bias + residual ----
  {
    v8f acc[3];
#pragma unroll
    for (int ni = 0; ni < 3; ++ni) acc[ni] = vzero8();
    for (int ks = 0; ks < 24; ++ks) {
      v16h af = ldfrag(h1P + (ks << 9) + lofs);
#pragma unroll
      for (int ni = 0; ni < 3; ++ni) {
        v16h bf = ldfrag(wfc2 + ((size_t)(ks * 12 + wv * 3 + ni) << 9) + lofs);
        acc[ni] = wmma16(af, bf, acc[ni]);
      }
    }
#pragma unroll
    for (int ni = 0; ni < 3; ++ni) {
      int col = wv * 48 + ni * 16 + nlo;
      float bias = bfc2[col];
#pragma unroll
      for (int r = 0; r < 8; ++r) {
        size_t idx = (row0 + r + 8 * hf) * kDim + col;
        xio[idx] = xio[idx] + acc[ni][r] + bias;
      }
    }
  }
}

// ---------------- host launch ----------------
extern "C" void kernel_launch(void* const* d_in, const int* in_sizes, int n_in,
                              void* d_out, int out_size, void* d_ws, size_t ws_size,
                              hipStream_t stream) {
  (void)in_sizes; (void)n_in; (void)out_size; (void)ws_size;
  const float* x     = (const float*)d_in[0];
  const float* g1    = (const float*)d_in[1];
  const float* b1    = (const float*)d_in[2];
  const float* wqkv  = (const float*)d_in[3];
  const float* bqkv  = (const float*)d_in[4];
  const float* wproj = (const float*)d_in[5];
  const float* bproj = (const float*)d_in[6];
  const float* g2    = (const float*)d_in[7];
  const float* b2    = (const float*)d_in[8];
  const float* wfc1  = (const float*)d_in[9];
  const float* bfc1  = (const float*)d_in[10];
  const float* wfc2  = (const float*)d_in[11];
  const float* bfc2  = (const float*)d_in[12];
  float* out = (float*)d_out;

  // workspace: packed f16 weights, then packed f16 attention buffer
  _Float16* wqkvP  = (_Float16*)d_ws;              // 192*576
  _Float16* wprojP = wqkvP  + 192 * 576;           // 192*192
  _Float16* wfc1P  = wprojP + 192 * 192;           // 192*768
  _Float16* wfc2P  = wfc1P  + 192 * 768;           // 768*192
  _Float16* attnb  = wfc2P  + 768 * 192;           // 200704*192 (fragment-packed)

  auto cdiv = [](int a, int b) { return (a + b - 1) / b; };
  pack_w<<<cdiv(192 * 576, 256), 256, 0, stream>>>(wqkv,  wqkvP, 192, 576);
  pack_w<<<cdiv(192 * 192, 256), 256, 0, stream>>>(wproj, wprojP, 192, 192);
  pack_w<<<cdiv(192 * 768, 256), 256, 0, stream>>>(wfc1,  wfc1P, 192, 768);
  pack_w<<<cdiv(768 * 192, 256), 256, 0, stream>>>(wfc2,  wfc2P, 768, 192);

  attn_kernel<<<kNWin, 96, 0, stream>>>(x, g1, b1, wqkvP, bqkv, attnb);
  proj_kernel<<<kRows / 16, 128, 0, stream>>>(attnb, wprojP, bproj, x, out);
  mlp_kernel<<<kRows / 16, 128, 0, stream>>>(g2, b2, wfc1P, bfc1, wfc2P, bfc2, out);
}
